// DecoderCell_56642028699792
// MI455X (gfx1250) — matrix operand order repaired and checked by reading, not compiled, not run on hardware
//
#include <hip/hip_runtime.h>
#include <hip/hip_bf16.h>
#include <math.h>

// ---------------------------------------------------------------------------
// Problem constants (fixed by the reference)
// ---------------------------------------------------------------------------
#define B_    2
#define L_    4096
#define D_    1024
#define H_    16
#define M_    256
#define DH_   64
#define LQ_   8192          // L * scale_factor(2)
#define EPS_    1e-6f
#define LN_EPS_ 1e-5f

typedef _Float16 h16;
typedef __attribute__((ext_vector_type(16))) _Float16 v16h;
typedef __attribute__((ext_vector_type(8)))  _Float16 v8h;
typedef __attribute__((ext_vector_type(8)))  float    v8f;

// ---------------------------------------------------------------------------
// WMMA helpers (CDNA5 wave32, 16x16x32 f16 -> f32)
// ---------------------------------------------------------------------------
__device__ __forceinline__ v8f wmma_f16(v16h a, v16h b, v8f c) {
    return __builtin_amdgcn_wmma_f32_16x16x32_f16(
        /*neg_a=*/false, a, /*neg_b=*/false, b,
        /*c_mod=*/(short)0, c, /*reuse_a=*/false, /*reuse_b=*/false);
}

// A fragment: 16x32 (MxK), row-major A, per ISA layout:
//   lane<16: k in {k0+0..7, k0+16..23}; lane>=16: k in {k0+8..15, k0+24..31}
// 16 contiguous-f16-in-pairs -> two 16B loads per lane.
__device__ __forceinline__ v16h load_a_frag(const h16* __restrict__ A, int lda,
                                            int row0, int k0, int rshift) {
    const int lane = threadIdx.x & 31;
    const int hf   = lane >> 4;
    const int m    = lane & 15;
    const h16* p = A + ((size_t)((row0 + m) >> rshift)) * (size_t)lda + k0 + hf * 8;
    v8h lo = *(const v8h*)(p);
    v8h hi = *(const v8h*)(p + 16);
    v16h a;
#pragma unroll
    for (int i = 0; i < 8; ++i) { a[i] = lo[i]; a[i + 8] = hi[i]; }
    return a;
}

// Same fragment layout but sourced from LDS (lowers to ds_load_b128).
__device__ __forceinline__ v16h load_a_frag_lds(const h16* A, int lda,
                                                int row0, int k0) {
    const int lane = threadIdx.x & 31;
    const int hf   = lane >> 4;
    const int m    = lane & 15;
    const h16* p = A + (size_t)(row0 + m) * (size_t)lda + k0 + hf * 8;
    v8h lo = *(const v8h*)(p);
    v8h hi = *(const v8h*)(p + 16);
    v16h a;
#pragma unroll
    for (int i = 0; i < 8; ++i) { a[i] = lo[i]; a[i + 8] = hi[i]; }
    return a;
}

// B fragment from a K-major ("Bt") matrix: Bt[n][k].
//   lane<16: k in {k0+0..7, k0+16..23}; lane>=16: k in {k0+8..15, k0+24..31}  (column n = lane&15)
__device__ __forceinline__ v16h load_b_frag(const h16* __restrict__ Bt, int ldb,
                                            int col0, int k0) {
    const int lane = threadIdx.x & 31;
    const int hf   = lane >> 4;
    const int n    = lane & 15;
    const h16* p = Bt + (size_t)(col0 + n) * (size_t)ldb + k0 + hf * 8;
    v8h lo = *(const v8h*)(p);
    v8h hi = *(const v8h*)(p + 16);
    v16h b;
#pragma unroll
    for (int i = 0; i < 8; ++i) { b[i] = lo[i]; b[i + 8] = hi[i]; }
    return b;
}

// ---------------------------------------------------------------------------
// CDNA5 async global->LDS copy (ASYNCcnt path). The low 32 bits of a flat
// shared-memory pointer are the LDS offset (aperture lives in addr[63:32]),
// so we can form the VDST operand directly from the shared-array address.
// ---------------------------------------------------------------------------
__device__ __forceinline__ void async_load_b128_to_lds(const void* gaddr, void* lds) {
    uint64_t g = (uint64_t)(uintptr_t)gaddr;
    uint32_t l = (uint32_t)(uintptr_t)lds;
    asm volatile("global_load_async_to_lds_b128 %0, %1, off"
                 :: "v"(l), "v"(g) : "memory");
}

__device__ __forceinline__ void wait_async_zero() {
#if __has_builtin(__builtin_amdgcn_s_wait_asynccnt)
    __builtin_amdgcn_s_wait_asynccnt(0);
#else
    asm volatile("s_wait_asynccnt 0" ::: "memory");
#endif
}

// ---------------------------------------------------------------------------
// Cast / transpose kernels
// ---------------------------------------------------------------------------
__global__ void k_cast_f16(const float* __restrict__ in, h16* __restrict__ out, int n) {
    int i = blockIdx.x * blockDim.x + threadIdx.x;
    if (i < n) out[i] = (h16)in[i];
}

// W [D,D] f32 row-major -> Wt[n][k] f16 (Wt[n*D+k] = W[k*D+n])
__global__ __launch_bounds__(256)
void k_cast_transpose(const float* __restrict__ W, h16* __restrict__ Wt) {
    __shared__ float tile[32][33];
    const int k0 = blockIdx.x * 32, n0 = blockIdx.y * 32;
    const int tx = threadIdx.x, ty = threadIdx.y;          // block (32, 8)
#pragma unroll
    for (int j = 0; j < 32; j += 8)
        tile[ty + j][tx] = W[(size_t)(k0 + ty + j) * D_ + n0 + tx];
    __syncthreads();
#pragma unroll
    for (int j = 0; j < 32; j += 8)
        Wt[(size_t)(n0 + ty + j) * D_ + k0 + tx] = (h16)tile[tx][ty + j];
}

// ---------------------------------------------------------------------------
// QKV projection GEMM: out[row, n] = scale * sum_k A[row>>rshift, k] * Bt[n, k]
// rshift=1 implements the nearest-neighbor sequence upsample for Q.
// mode 0: store [rows, D] f16.  mode 1: store V transposed per head:
//         vT[((b*H + h)*DH + d) * L + l]
// One 32x32 tile per wave, 8 waves per block.
// ---------------------------------------------------------------------------
__global__ __launch_bounds__(256)
void k_gemm_qkv(const h16* __restrict__ A, const h16* __restrict__ Bt,
                h16* __restrict__ out, int rows, int rshift, float scale, int mode) {
    const int K = D_, N = D_;
    const int lane = threadIdx.x & 31;
    const int wave = threadIdx.x >> 5;
    const int tiles_n = N / 32;
    const int tid = blockIdx.x * 8 + wave;
    const int tm = tid / tiles_n, tn = tid % tiles_n;
    const int row0 = tm * 32, col0 = tn * 32;
    if (row0 >= rows) return;

    v8f acc[2][2] = {};
    for (int k0 = 0; k0 < K; k0 += 32) {
        if (k0 + 128 < K) {  // L2 prefetch hints (global_prefetch_b8)
            __builtin_prefetch(A + ((size_t)((row0 + (lane & 15)) >> rshift)) * K + k0 + 128, 0, 1);
            __builtin_prefetch(Bt + (size_t)(col0 + (lane & 15)) * K + k0 + 128, 0, 1);
        }
        v16h a0 = load_a_frag(A, K, row0,      k0, rshift);
        v16h a1 = load_a_frag(A, K, row0 + 16, k0, rshift);
        v16h b0 = load_b_frag(Bt, K, col0,      k0);
        v16h b1 = load_b_frag(Bt, K, col0 + 16, k0);
        acc[0][0] = wmma_f16(a0, b0, acc[0][0]);
        acc[0][1] = wmma_f16(a0, b1, acc[0][1]);
        acc[1][0] = wmma_f16(a1, b0, acc[1][0]);
        acc[1][1] = wmma_f16(a1, b1, acc[1][1]);
    }

    const int hf = lane >> 4, n = lane & 15;
    if (mode == 0) {
#pragma unroll
        for (int mi = 0; mi < 2; ++mi)
#pragma unroll
            for (int ni = 0; ni < 2; ++ni)
#pragma unroll
                for (int i = 0; i < 8; ++i) {
                    int r = row0 + mi * 16 + hf * 8 + i;
                    int c = col0 + ni * 16 + n;
                    out[(size_t)r * N + c] = (h16)(acc[mi][ni][i] * scale);
                }
    } else {
        // V transposed per head; rows index [0, B*L)
#pragma unroll
        for (int mi = 0; mi < 2; ++mi)
#pragma unroll
            for (int ni = 0; ni < 2; ++ni) {
                int c = col0 + ni * 16 + n;
                int h = c >> 6, d = c & 63;
#pragma unroll
                for (int i = 0; i < 8; ++i) {
                    int r = row0 + mi * 16 + hf * 8 + i;
                    int b = r >> 12;           // / L_
                    int l = r & (L_ - 1);
                    out[((size_t)(b * H_ + h) * DH_ + d) * L_ + l] = (h16)acc[mi][ni][i];
                }
            }
    }
}

// ---------------------------------------------------------------------------
// FAVOR+ feature map. One block (4 waves, 128 thr) computes a 32-row x 256-col
// "dash" tile for one (b,h): dash = u @ proj[h]^T via WMMA (K=DH=64), then
// qf = exp(dash - 0.5*|u|^2 - rowmax) * M^-0.5.
// kfT_mode=1 stores transposed [b,h,m,L] (feeds the kv GEMM as A).
// ---------------------------------------------------------------------------
__global__ __launch_bounds__(128)
void k_favor(const h16* __restrict__ feat,  // [B*Lrows, D] f16 (already scaled)
             const h16* __restrict__ proj,  // [H, M, DH] f16  (== Bt layout)
             h16* __restrict__ out, int Lrows, int kfT_mode) {
    __shared__ float dash[32][M_];
    __shared__ float diag_s[32];
    __shared__ float part[32][4];
    __shared__ float rowstat[32];

    const int tidx = threadIdx.x;
    const int lane = tidx & 31;
    const int wave = tidx >> 5;
    const int bh = blockIdx.y;
    const int b = bh / H_, h = bh % H_;
    const int row0 = blockIdx.x * 32;

    const h16* A  = feat + ((size_t)b * Lrows + row0) * D_ + h * DH_;
    const h16* Bt = proj + (size_t)h * M_ * DH_;

    if (tidx < 32) {  // diag = 0.5 * sum_d u^2 per row
        float s = 0.f;
        const h16* p = A + (size_t)tidx * D_;
#pragma unroll
        for (int d = 0; d < DH_; ++d) { float v = (float)p[d]; s += v * v; }
        diag_s[tidx] = 0.5f * s;
    }

    v8f acc[2][4] = {};
#pragma unroll
    for (int ks = 0; ks < 2; ++ks) {
        const int k0 = ks * 32;
        v16h a0 = load_a_frag(A, D_, 0,  k0, 0);
        v16h a1 = load_a_frag(A, D_, 16, k0, 0);
#pragma unroll
        for (int nt = 0; nt < 4; ++nt) {
            v16h bf = load_b_frag(Bt, DH_, wave * 64 + nt * 16, k0);
            acc[0][nt] = wmma_f16(a0, bf, acc[0][nt]);
            acc[1][nt] = wmma_f16(a1, bf, acc[1][nt]);
        }
    }
    const int hf = lane >> 4, n = lane & 15;
#pragma unroll
    for (int mi = 0; mi < 2; ++mi)
#pragma unroll
        for (int nt = 0; nt < 4; ++nt)
#pragma unroll
            for (int i = 0; i < 8; ++i)
                dash[mi * 16 + hf * 8 + i][wave * 64 + nt * 16 + n] = acc[mi][nt][i];
    __syncthreads();

    {   // row max over the 256 columns: 4 threads x 64 cols per row
        int r = tidx >> 2, seg = tidx & 3;
        float mx = -1e30f;
        const float* dr = &dash[r][seg * 64];
        for (int j = 0; j < 64; ++j) mx = fmaxf(mx, dr[j]);
        part[r][seg] = mx;
    }
    __syncthreads();
    if (tidx < 32) {
        float mx = fmaxf(fmaxf(part[tidx][0], part[tidx][1]),
                         fmaxf(part[tidx][2], part[tidx][3]));
        rowstat[tidx] = mx + diag_s[tidx];
    }
    __syncthreads();

    const float inv_sqrt_m = 0.0625f;  // 256^-0.5
    for (int idx = tidx; idx < 32 * M_; idx += 128) {
        int r = idx >> 8, m = idx & 255;
        float v = expf(dash[r][m] - rowstat[r]) * inv_sqrt_m;
        size_t l = (size_t)row0 + r;
        if (!kfT_mode)
            out[((size_t)bh * Lrows + l) * M_ + m] = (h16)v;           // qf [b,h,l,m]
        else
            out[((size_t)bh * M_ + m) * L_ + l] = (h16)v;              // kfT [b,h,m,l]
    }
}

// ---------------------------------------------------------------------------
// k_sum[b,h,m] = sum_l kfT[b,h,m,l]  (one wave per m-row)
// ---------------------------------------------------------------------------
__global__ __launch_bounds__(256)
void k_ksum(const h16* __restrict__ kfT, float* __restrict__ ksum) {
    const int lane = threadIdx.x & 31, wave = threadIdx.x >> 5;
    const int row = blockIdx.x * 8 + wave;  // [0, B*H*M)
    const h16* p = kfT + (size_t)row * L_;
    float s = 0.f;
#pragma unroll 4
    for (int j = 0; j < L_ / 256; ++j) {
        v8h v = *(const v8h*)(p + (j * 32 + lane) * 8);
#pragma unroll
        for (int e = 0; e < 8; ++e) s += (float)v[e];
    }
#pragma unroll
    for (int off = 16; off > 0; off >>= 1) s += __shfl_xor(s, off, 32);
    if (lane == 0) ksum[row] = s;
}

// ---------------------------------------------------------------------------
// kv GEMM per (b,h): kv[m,d] = sum_l kfT[m,l] * vT[d,l]  (K = L = 4096)
// Stored transposed: kvT[b,h,d,m] so it serves as Bt for the attn GEMM.
// ---------------------------------------------------------------------------
__global__ __launch_bounds__(256)
void k_gemm_kv(const h16* __restrict__ kfT, const h16* __restrict__ vT,
               h16* __restrict__ kvT) {
    const int lane = threadIdx.x & 31, wave = threadIdx.x >> 5;
    const int bh = blockIdx.y;
    const h16* A  = kfT + (size_t)bh * M_ * L_;   // [M, L]
    const h16* Bt = vT  + (size_t)bh * DH_ * L_;  // [DH, L]
    h16* C = kvT + (size_t)bh * DH_ * M_;         // [DH, M]

    const int tid = blockIdx.x * 8 + wave;  // 16 tiles per (b,h)
    const int tm = tid >> 1, tn = tid & 1;
    const int row0 = tm * 32, col0 = tn * 32;

    v8f acc[2][2] = {};
    for (int k0 = 0; k0 < L_; k0 += 32) {
        if (k0 + 128 < L_) {
            __builtin_prefetch(A + (size_t)(row0 + (lane & 15)) * L_ + k0 + 128, 0, 1);
            __builtin_prefetch(Bt + (size_t)(col0 + (lane & 15)) * L_ + k0 + 128, 0, 1);
        }
        v16h a0 = load_a_frag(A, L_, row0,      k0, 0);
        v16h a1 = load_a_frag(A, L_, row0 + 16, k0, 0);
        v16h b0 = load_b_frag(Bt, L_, col0,      k0);
        v16h b1 = load_b_frag(Bt, L_, col0 + 16, k0);
        acc[0][0] = wmma_f16(a0, b0, acc[0][0]);
        acc[0][1] = wmma_f16(a0, b1, acc[0][1]);
        acc[1][0] = wmma_f16(a1, b0, acc[1][0]);
        acc[1][1] = wmma_f16(a1, b1, acc[1][1]);
    }
    const int hf = lane >> 4, n = lane & 15;
#pragma unroll
    for (int mi = 0; mi < 2; ++mi)
#pragma unroll
        for (int ni = 0; ni < 2; ++ni) {
            int d = col0 + ni * 16 + n;
#pragma unroll
            for (int i = 0; i < 8; ++i) {
                int m = row0 + mi * 16 + hf * 8 + i;
                C[(size_t)d * M_ + m] = (h16)acc[mi][ni][i];
            }
        }
}

// ---------------------------------------------------------------------------
// attn per (b,h): attn[l,d] = (qf[l,:] @ kv[:,d]) * z[l],
//                 z[l] = 1/(qf[l,:] . k_sum + eps)
// Each wave does 32 rows x full DH=64 cols (K = M = 256); the z dot-product
// is accumulated alongside from the A fragments + ksum (f32).
// Output written as [B*Lq, D] f16 with per-head column offset -> Wo GEMM A.
// ---------------------------------------------------------------------------
__global__ __launch_bounds__(256)
void k_attn(const h16* __restrict__ qf, const h16* __restrict__ kvT,
            const float* __restrict__ ksum, h16* __restrict__ attn) {
    const int lane = threadIdx.x & 31, wave = threadIdx.x >> 5;
    const int bh = blockIdx.y;
    const int b = bh >> 4, h = bh & 15;
    const h16* A  = qf  + (size_t)bh * LQ_ * M_;   // [Lq, M]
    const h16* Bt = kvT + (size_t)bh * DH_ * M_;   // [DH, M]
    const float* ks = ksum + (size_t)bh * M_;

    const int row0 = (blockIdx.x * 8 + wave) * 32;
    const int hf = lane >> 4, n = lane & 15;

    v8f acc[2][4] = {};
    float za0 = 0.f, za1 = 0.f;
    for (int k0 = 0; k0 < M_; k0 += 32) {
        v16h a0 = load_a_frag(A, M_, row0,      k0, 0);
        v16h a1 = load_a_frag(A, M_, row0 + 16, k0, 0);
        const float* kp = ks + k0 + hf * 8;
#pragma unroll
        for (int j = 0; j < 8; ++j) {
            float klo = kp[j], khi = kp[16 + j];
            za0 += (float)a0[j] * klo + (float)a0[j + 8] * khi;
            za1 += (float)a1[j] * klo + (float)a1[j + 8] * khi;
        }
#pragma unroll
        for (int nt = 0; nt < 4; ++nt) {
            v16h bf = load_b_frag(Bt, M_, nt * 16, k0);
            acc[0][nt] = wmma_f16(a0, bf, acc[0][nt]);
            acc[1][nt] = wmma_f16(a1, bf, acc[1][nt]);
        }
    }
    // combine the two K-halves of each row (lanes r and r+16 hold row r)
    za0 += __shfl_xor(za0, 16, 32);
    za1 += __shfl_xor(za1, 16, 32);

#pragma unroll
    for (int mi = 0; mi < 2; ++mi) {
        float zam = (mi == 0) ? za0 : za1;
#pragma unroll
        for (int i = 0; i < 8; ++i) {
            float den = __shfl(zam, hf * 8 + i, 32);
            float z = 1.0f / (den + EPS_);
            int l = row0 + mi * 16 + hf * 8 + i;
            size_t row = (size_t)b * LQ_ + l;
#pragma unroll
            for (int nt = 0; nt < 4; ++nt) {
                int d = nt * 16 + n;
                attn[row * D_ + h * DH_ + d] = (h16)(acc[mi][nt][i] * z);
            }
        }
    }
}

// ---------------------------------------------------------------------------
// y = attn @ Wo, exact GELU, LayerNorm over D=1024, write f32 output.
// One block (8 waves) owns 16 full rows. The 16x1024 f16 A-panel is shared by
// all 8 waves, so it is staged ONCE into LDS via CDNA5 async global->LDS
// copies (ASYNCcnt path, no VGPR round-trip); WMMA A-fragments then come from
// LDS while B-fragments stream from global. y stays in registers; cross-wave
// mean/var via shuffle + small LDS reduction.
// ---------------------------------------------------------------------------
__global__ __launch_bounds__(256)
void k_out(const h16* __restrict__ attn, const h16* __restrict__ woT,
           const float* __restrict__ lng, const float* __restrict__ lnb,
           float* __restrict__ out) {
    __shared__ h16 sA[16 * D_];          // 32 KiB staged A-panel
    __shared__ float red[8][16][2];
    __shared__ float mus[16], rss[16];

    const int lane = threadIdx.x & 31, wave = threadIdx.x >> 5;
    const int row0 = blockIdx.x * 16;
    const int colw = wave * 128;
    const int hf = lane >> 4, n = lane & 15;

    // Stage the block's 16-row x 1024-col A panel: 2048 x 16B async copies.
    {
        const h16* gbase = attn + (size_t)row0 * D_;
#pragma unroll
        for (int j = 0; j < 8; ++j) {
            int idx = threadIdx.x + j * 256;   // [0, 2048) 16-byte chunks
            async_load_b128_to_lds(gbase + (size_t)idx * 8, &sA[idx * 8]);
        }
        wait_async_zero();
        __syncthreads();
    }

    v8f acc[8] = {};
    for (int k0 = 0; k0 < D_; k0 += 32) {
        v16h a = load_a_frag_lds(sA, D_, 0, k0);   // ds_load_b128 from LDS
#pragma unroll
        for (int nt = 0; nt < 8; ++nt) {
            v16h bf = load_b_frag(woT, D_, colw + nt * 16, k0);
            acc[nt] = wmma_f16(a, bf, acc[nt]);
        }
    }

    float gy[8][8];
    float psum[8], psq[8];
#pragma unroll
    for (int i = 0; i < 8; ++i) { psum[i] = 0.f; psq[i] = 0.f; }
#pragma unroll
    for (int nt = 0; nt < 8; ++nt)
#pragma unroll
        for (int i = 0; i < 8; ++i) {
            float x = acc[nt][i];
            float g = 0.5f * x * (1.0f + erff(x * 0.70710678118654752f));  // exact GELU
            gy[nt][i] = g;
            psum[i] += g;
            psq[i]  += g * g;
        }
#pragma unroll
    for (int off = 1; off < 16; off <<= 1) {
#pragma unroll
        for (int i = 0; i < 8; ++i) {
            psum[i] += __shfl_xor(psum[i], off, 32);
            psq[i]  += __shfl_xor(psq[i],  off, 32);
        }
    }
    if (n == 0) {
#pragma unroll
        for (int i = 0; i < 8; ++i) {
            red[wave][hf * 8 + i][0] = psum[i];
            red[wave][hf * 8 + i][1] = psq[i];
        }
    }
    __syncthreads();
    if (threadIdx.x < 16) {
        float s = 0.f, s2 = 0.f;
#pragma unroll
        for (int w = 0; w < 8; ++w) { s += red[w][threadIdx.x][0]; s2 += red[w][threadIdx.x][1]; }
        float mu  = s * (1.0f / D_);
        float var = s2 * (1.0f / D_) - mu * mu;
        mus[threadIdx.x] = mu;
        rss[threadIdx.x] = rsqrtf(var + LN_EPS_);
    }
    __syncthreads();
#pragma unroll
    for (int nt = 0; nt < 8; ++nt) {
        int c = colw + nt * 16 + n;
        float gg = lng[c], bb = lnb[c];
#pragma unroll
        for (int i = 0; i < 8; ++i) {
            int r = hf * 8 + i;
            out[(size_t)(row0 + r) * D_ + c] = (gy[nt][i] - mus[r]) * rss[r] * gg + bb;
        }
    }
}

// ---------------------------------------------------------------------------
// Host-side launcher
// ---------------------------------------------------------------------------
extern "C" void kernel_launch(void* const* d_in, const int* in_sizes, int n_in,
                              void* d_out, int out_size, void* d_ws, size_t ws_size,
                              hipStream_t stream) {
    (void)in_sizes; (void)n_in; (void)out_size; (void)ws_size;

    const float* x    = (const float*)d_in[0];
    const float* Wq   = (const float*)d_in[1];
    const float* Wk   = (const float*)d_in[2];
    const float* Wv   = (const float*)d_in[3];
    const float* Wo   = (const float*)d_in[4];
    const float* proj = (const float*)d_in[5];
    const float* ln_g = (const float*)d_in[6];
    const float* ln_b = (const float*)d_in[7];
    float* out = (float*)d_out;

    char* ws = (char*)d_ws;
    size_t off = 0;
    auto take = [&](size_t bytes) -> void* {
        void* p = ws + off;
        off += (bytes + 255) & ~(size_t)255;
        return p;
    };
    h16* x16    = (h16*)take((size_t)B_ * L_ * D_ * 2);       // 16 MiB
    h16* wqT    = (h16*)take((size_t)D_ * D_ * 2);
    h16* wkT    = (h16*)take((size_t)D_ * D_ * 2);
    h16* wvT    = (h16*)take((size_t)D_ * D_ * 2);
    h16* woT    = (h16*)take((size_t)D_ * D_ * 2);
    h16* proj16 = (h16*)take((size_t)H_ * M_ * DH_ * 2);
    h16* q16    = (h16*)take((size_t)B_ * LQ_ * D_ * 2);      // 32 MiB
    h16* k16    = (h16*)take((size_t)B_ * L_ * D_ * 2);       // 16 MiB
    h16* vT16   = (h16*)take((size_t)B_ * H_ * DH_ * L_ * 2); // 16 MiB
    h16* qf16   = (h16*)take((size_t)B_ * H_ * LQ_ * M_ * 2); // 128 MiB
    h16* kfT16  = (h16*)take((size_t)B_ * H_ * M_ * L_ * 2);  // 64 MiB
    h16* kvT16  = (h16*)take((size_t)B_ * H_ * DH_ * M_ * 2); // 1 MiB
    float* ksum = (float*)take((size_t)B_ * H_ * M_ * 4);
    h16* attn16 = (h16*)take((size_t)B_ * LQ_ * D_ * 2);      // 32 MiB

    const float qk_scale = 0.35355339059327373f;  // DH^-0.25 = 64^-0.25

    // 1) casts
    k_cast_f16<<<(B_ * L_ * D_ + 255) / 256, 256, 0, stream>>>(x, x16, B_ * L_ * D_);
    k_cast_f16<<<(H_ * M_ * DH_ + 255) / 256, 256, 0, stream>>>(proj, proj16, H_ * M_ * DH_);
    {
        dim3 tb(32, 8), tg(D_ / 32, D_ / 32);
        k_cast_transpose<<<tg, tb, 0, stream>>>(Wq, wqT);
        k_cast_transpose<<<tg, tb, 0, stream>>>(Wk, wkT);
        k_cast_transpose<<<tg, tb, 0, stream>>>(Wv, wvT);
        k_cast_transpose<<<tg, tb, 0, stream>>>(Wo, woT);
    }

    // 2) QKV projections (q rows read up[l] = x[l/2] via rshift=1)
    {
        int rows_q = B_ * LQ_;
        int blocks_q = (rows_q / 32) * (D_ / 32) / 8;
        k_gemm_qkv<<<blocks_q, 256, 0, stream>>>(x16, wqT, q16, rows_q, 1, qk_scale, 0);
        int rows_kv = B_ * L_;
        int blocks_kv = (rows_kv / 32) * (D_ / 32) / 8;
        k_gemm_qkv<<<blocks_kv, 256, 0, stream>>>(x16, wkT, k16, rows_kv, 0, qk_scale, 0);
        k_gemm_qkv<<<blocks_kv, 256, 0, stream>>>(x16, wvT, vT16, rows_kv, 0, 1.0f, 1);
    }

    // 3) FAVOR features
    k_favor<<<dim3(LQ_ / 32, B_ * H_), 128, 0, stream>>>(q16, proj16, qf16, LQ_, 0);
    k_favor<<<dim3(L_ / 32,  B_ * H_), 128, 0, stream>>>(k16, proj16, kfT16, L_, 1);

    // 4) k_sum
    k_ksum<<<(B_ * H_ * M_) / 8, 256, 0, stream>>>(kfT16, ksum);

    // 5) kv = kf^T @ v  -> kvT
    k_gemm_kv<<<dim3(2, B_ * H_), 256, 0, stream>>>(kfT16, vT16, kvT16);

    // 6) attn = (qf @ kv) * z
    k_attn<<<dim3(LQ_ / 32 / 8, B_ * H_), 256, 0, stream>>>(qf16, kvT16, ksum, attn16);

    // 7) y = attn @ Wo, GELU, LayerNorm -> out (f32)
    k_out<<<(B_ * LQ_) / 16, 256, 0, stream>>>(attn16, woT, ln_g, ln_b, out);
}
